// EfficientAttention_36507222016383
// MI455X (gfx1250) — compile-verified, matrix operand order checked by reading, not run on hardware
//
#include <hip/hip_runtime.h>
#include <hip/hip_bf16.h>
#include <stdint.h>

#define B_SZ  16
#define CH    256
#define LSEQ  8192
#define HEADS_N 8
#define HD    32

typedef __attribute__((ext_vector_type(16))) __bf16 v16bf;
typedef __attribute__((ext_vector_type(8)))  float  v8f;
typedef __attribute__((ext_vector_type(4)))  unsigned int u32x4;
typedef __attribute__((ext_vector_type(8)))  int i32x8;
typedef __attribute__((ext_vector_type(4)))  int i32x4;

#if defined(__has_builtin)
#if __has_builtin(__builtin_amdgcn_tensor_load_to_lds) && __has_builtin(__builtin_amdgcn_s_wait_tensorcnt)
#define USE_TDM 1
#endif
#endif
#ifndef USE_TDM
#define USE_TDM 0
#endif

// ---- bf16 <-> f32 helpers (explicit RNE, no reliance on __bf16 cast lowering) ----
static __device__ inline __bf16 f2bf(float f) {
    union { float f; unsigned u; } in; in.f = f;
    unsigned r = (in.u + 0x7FFFu + ((in.u >> 16) & 1u)) >> 16;
    union { unsigned short s; __bf16 b; } cv; cv.s = (unsigned short)r;
    return cv.b;
}
static __device__ inline float bf2f(__bf16 b) {
    union { unsigned short s; __bf16 b; } cv; cv.b = b;
    union { unsigned u; float f; } out; out.u = ((unsigned)cv.s) << 16;
    return out.f;
}

// ---- WMMA 16x16x32 bf16 fragment loader (wave32 ISA layout) ----
// Each lane's 16 bf16 values are two contiguous 8-element runs at K = kh*8 and
// K = 16 + kh*8  ->  two 16-byte loads. B fragments use the same loader on B^T.
static __device__ inline v16bf load_frag(const __bf16* base, int stride_elems) {
    int lane = threadIdx.x & 31;
    const __bf16* row = base + (size_t)(lane & 15) * stride_elems + ((lane >> 4) << 3);
    union { v16bf v; uint4 q[2]; } f;
    f.q[0] = *(const uint4*)(row);
    f.q[1] = *(const uint4*)(row + 16);
    return f.v;
}

static __device__ inline v8f wmma_bf16(v16bf a, v16bf b, v8f c) {
    return __builtin_amdgcn_wmma_f32_16x16x32_bf16(false, a, false, b, (short)0, c, false, false);
}

#if USE_TDM
// ---- Tensor Data Mover: 2D tile (tile_x contiguous elems, tile_y rows with
// row stride stride_x elems) from global -> contiguous LDS at lds_off.
// D# per CDNA5 ISA: group0 = {count=1, lds_addr,
// global_addr[56:0], type=2}; group1 = {data_size, tensor_dims = tile extents
// (no OOB clipping), tile_dims, tensor_dim0_stride}. Groups 2/3 unused (2D).
// This toolchain exposes the 6-arg builtin: (g0, g1, g2, g3, g_extra, cpol).
static __device__ inline void tdm_load_2d(unsigned lds_off, unsigned long long ga,
                                          int dsz /*0=1B,1=2B,2=4B*/,
                                          int tile_x, int tile_y, long long stride_x) {
    u32x4 g0;
    g0[0] = 1u;                                                   // count=1 (valid)
    g0[1] = lds_off;                                              // lds_addr
    g0[2] = (unsigned)ga;                                         // global_addr lo
    g0[3] = (unsigned)((ga >> 32) & 0x1FFFFFFull) | (2u << 30);   // addr hi | type=2
    i32x8 g1;
    g1[0] = dsz << 16;                                   // workgroup_mask=0, data_size
    g1[1] = (int)(((unsigned)tile_x & 0xFFFFu) << 16);   // tensor_dim0 lo16
    g1[2] = (int)(((unsigned)tile_y & 0xFFFFu) << 16);   // dim0 hi=0 | tensor_dim1 lo16
    g1[3] = (int)(((unsigned)tile_x) << 16);             // dim1 hi=0 | tile_dim0
    g1[4] = tile_y;                                      // tile_dim1 | tile_dim2=0
    g1[5] = (int)(stride_x & 0xFFFFFFFFll);              // tensor_dim0_stride lo32
    g1[6] = (int)((stride_x >> 32) & 0xFFFFll);          // stride0 hi16 | stride1 lo=0
    g1[7] = 0;
    i32x4 gz4 = {0, 0, 0, 0};
    i32x8 gz8 = {0, 0, 0, 0, 0, 0, 0, 0};
    __builtin_amdgcn_tensor_load_to_lds(g0, g1, gz4, gz4, gz8, 0);
}
#endif

// =====================================================================
// Kernel 0: weights -> bf16, stacked bias
// =====================================================================
__global__ __launch_bounds__(256)
void prep_kernel(const float* __restrict__ Wk, const float* __restrict__ Wq,
                 const float* __restrict__ Wv, const float* __restrict__ Wr,
                 const float* __restrict__ bk, const float* __restrict__ bq,
                 const float* __restrict__ bv,
                 __bf16* __restrict__ wkqv, __bf16* __restrict__ wr,
                 float* __restrict__ biasS) {
    int idx = blockIdx.x * 256 + threadIdx.x;
    if (idx < 768 * 256) {
        int o = idx >> 8, c = idx & 255;
        float v = (o < 256) ? Wk[o * 256 + c]
                : (o < 512) ? Wq[(o - 256) * 256 + c]
                            : Wv[(o - 512) * 256 + c];
        wkqv[idx] = f2bf(v);
    } else if (idx < 768 * 256 + 65536) {
        int j = idx - 768 * 256;
        wr[j] = f2bf(Wr[j]);
    } else if (idx < 768 * 256 + 65536 + 768) {
        int j = idx - (768 * 256 + 65536);
        biasS[j] = (j < 256) ? bk[j] : (j < 512) ? bq[j - 256] : bv[j - 512];
    }
}

// =====================================================================
// Kernel 1: fused K/Q/V projection. Stacked 768x64 output tile per block.
// Stores exp(k + bk) directly for the K third. Input tile staged by TDM.
// kqv layout: [sel(0=E,1=Q,2=V)][B][CH][LSEQ]
// =====================================================================
__global__ __launch_bounds__(256)
void proj_kernel(const float* __restrict__ x, const __bf16* __restrict__ wkqv,
                 const float* __restrict__ biasS, __bf16* __restrict__ kqv) {
    __shared__ __align__(16) __bf16 xT[64 * 264];   // [l][c], padded stride
#if USE_TDM
    __shared__ __align__(16) float xF32[64 * 256];  // [c][l] rows, TDM destination
#endif
    int b  = blockIdx.y;
    int l0 = blockIdx.x * 64;
    int tid = threadIdx.x;

    __builtin_prefetch(wkqv + (tid << 6), 0, 1);    // pull weights toward L2

#if USE_TDM
    if (tid < 32) {   // one wave issues the DMA (TDM ignores EXEC; branch-guarded)
        tdm_load_2d((unsigned)(uintptr_t)&xF32[0],
                    (unsigned long long)(uintptr_t)(x + (size_t)(b * CH) * LSEQ + l0),
                    /*dsz f32*/2, /*tile_x*/64, /*tile_y*/CH, /*stride*/LSEQ);
        __builtin_amdgcn_s_wait_tensorcnt(0);
    }
    __syncthreads();
    // transpose + convert out of LDS: [c][l] f32 -> [l][c] bf16
    for (int i = tid; i < 64 * 64; i += 256) {
        int c = i >> 4;
        int l = (i & 15) * 4;
        float4 v = *(const float4*)&xF32[c * 64 + l];
        xT[(l + 0) * 264 + c] = f2bf(v.x);
        xT[(l + 1) * 264 + c] = f2bf(v.y);
        xT[(l + 2) * 264 + c] = f2bf(v.z);
        xT[(l + 3) * 264 + c] = f2bf(v.w);
    }
#else
    for (int i = tid; i < 64 * 128; i += 256) {
        int c = i >> 5;
        int l = (i & 31) * 2;
        const float* src = x + ((size_t)(b * CH + c)) * LSEQ + l0 + l;
        float2 v = *(const float2*)src;
        xT[l * 264 + c]       = f2bf(v.x);
        xT[(l + 1) * 264 + c] = f2bf(v.y);
    }
#endif
    __syncthreads();

    int wave = tid >> 5, lane = tid & 31, kh = lane >> 4;
    for (int t = 0; t < 24; ++t) {
        int tile = wave + (t << 3);     // 192 tiles = 48 row-tiles x 4 l-tiles
        int rt = tile >> 2;
        int lt = tile & 3;
        v8f acc = {};
        #pragma unroll
        for (int kk = 0; kk < 8; ++kk) {
            v16bf a  = load_frag(wkqv + (size_t)(rt * 16) * CH + kk * 32, CH);
            v16bf bb = load_frag(xT + (lt * 16) * 264 + kk * 32, 264);
            acc = wmma_bf16(a, bb, acc);
        }
        int sel = rt >> 4;                          // 0=K,1=Q,2=V (uniform per tile)
        int l   = l0 + lt * 16 + (lane & 15);
        #pragma unroll
        for (int r = 0; r < 8; ++r) {
            int row = rt * 16 + kh * 8 + r;         // C layout: M = r + 8*kh
            float val = acc[r] + biasS[row];
            if (sel == 0) val = __expf(val);
            int co = row & 255;
            kqv[(((size_t)sel * B_SZ + b) * CH + co) * LSEQ + l] = f2bf(val);
        }
    }
}

// =====================================================================
// Kernel 2: context per (b,h): C[i][j] = sum_l E[i,l]*V[j,l], Z[i] = sum_l E[i,l]
// One block per (b,h). 8 waves split K=8192; deterministic LDS reduction.
// =====================================================================
__global__ __launch_bounds__(256)
void context_kernel(const __bf16* __restrict__ kqv, __bf16* __restrict__ cn) {
    __shared__ __align__(16) float cbufW[8][1024];
    __shared__ float zpart[32][8];
    __shared__ float zfin[32];
    int bh = blockIdx.x;
    int b = bh >> 3, h = bh & 7;
    int tid = threadIdx.x, wave = tid >> 5, lane = tid & 31;
    const __bf16* E = kqv + ((size_t)b * CH + h * HD) * LSEQ;
    const __bf16* V = kqv + (((size_t)2 * B_SZ + b) * CH + h * HD) * LSEQ;

    v8f a00 = {}, a01 = {}, a10 = {}, a11 = {};
    for (int it = 0; it < 32; ++it) {
        int l = (wave + (it << 3)) << 5;
        v16bf fa0 = load_frag(E + l, LSEQ);
        v16bf fa1 = load_frag(E + 16 * LSEQ + l, LSEQ);
        v16bf fb0 = load_frag(V + l, LSEQ);
        v16bf fb1 = load_frag(V + 16 * LSEQ + l, LSEQ);
        a00 = wmma_bf16(fa0, fb0, a00);
        a01 = wmma_bf16(fa0, fb1, a01);
        a10 = wmma_bf16(fa1, fb0, a10);
        a11 = wmma_bf16(fa1, fb1, a11);
    }
    {
        int n0 = lane & 15, moff = (lane >> 4) * 8;
        #pragma unroll
        for (int r = 0; r < 8; ++r) {
            cbufW[wave][(moff + r) * 32 + n0]            = a00[r];
            cbufW[wave][(moff + r) * 32 + 16 + n0]       = a01[r];
            cbufW[wave][(16 + moff + r) * 32 + n0]       = a10[r];
            cbufW[wave][(16 + moff + r) * 32 + 16 + n0]  = a11[r];
        }
    }
    {   // Z: fixed-order row sums
        int row = tid & 31, part = tid >> 5;
        const __bf16* p = E + (size_t)row * LSEQ + part * 1024;
        float s = 0.f;
        for (int i = 0; i < 128; ++i) {
            uint4 q = ((const uint4*)p)[i];
            const __bf16* e = (const __bf16*)&q;
            #pragma unroll
            for (int j = 0; j < 8; ++j) s += bf2f(e[j]);
        }
        zpart[row][part] = s;
    }
    __syncthreads();
    if (tid < 32) {
        float s = 0.f;
        #pragma unroll
        for (int w = 0; w < 8; ++w) s += zpart[tid][w];
        zfin[tid] = s;
    }
    __syncthreads();
    for (int idx = tid; idx < 1024; idx += 256) {
        float s = 0.f;
        #pragma unroll
        for (int w = 0; w < 8; ++w) s += cbufW[w][idx];
        int i = idx >> 5, j = idx & 31;
        cn[(size_t)bh * 1024 + j * 32 + i] = f2bf(s / zfin[i]);   // C^T / Z
    }
}

// =====================================================================
// Kernel 3: Q tile staged by TDM -> q-softmax -> attended = Cn^T @ qn per head
// -> out = Wr @ attended + br + x.
// =====================================================================
__global__ __launch_bounds__(256)
void output_kernel(const float* __restrict__ x, const __bf16* __restrict__ kqv,
                   const __bf16* __restrict__ cn, const __bf16* __restrict__ wr,
                   const float* __restrict__ br, float* __restrict__ out) {
    __shared__ __align__(16) __bf16 qn[64 * 264];    // [l][c]
    __shared__ __align__(16) __bf16 att[64 * 264];   // [l][v]
#if USE_TDM
    __shared__ __align__(16) __bf16 qF[256 * 64];    // [c][l], TDM destination
#endif
    int b  = blockIdx.y;
    int l0 = blockIdx.x * 64;
    int tid = threadIdx.x, wave = tid >> 5, lane = tid & 31, kh = lane >> 4;
    const __bf16* Q = kqv + (((size_t)B_SZ + b) * CH) * LSEQ;

    __builtin_prefetch(wr + (tid << 6), 0, 1);

#if USE_TDM
    if (tid < 32) {
        tdm_load_2d((unsigned)(uintptr_t)&qF[0],
                    (unsigned long long)(uintptr_t)(Q + l0),
                    /*dsz bf16*/1, /*tile_x*/64, /*tile_y*/CH, /*stride*/LSEQ);
        __builtin_amdgcn_s_wait_tensorcnt(0);
    }
    __syncthreads();
#endif

    // q softmax over the 32 channels of each (head, l)
    for (int p = tid; p < 512; p += 256) {
        int h = p >> 6, l = p & 63;
        float vals[32];
        float mx = -1e30f;
        #pragma unroll
        for (int c = 0; c < 32; ++c) {
#if USE_TDM
            vals[c] = bf2f(qF[(h * 32 + c) * 64 + l]);
#else
            vals[c] = bf2f(Q[((size_t)(h * 32 + c)) * LSEQ + l0 + l]);
#endif
            mx = fmaxf(mx, vals[c]);
        }
        float s = 0.f;
        #pragma unroll
        for (int c = 0; c < 32; ++c) { vals[c] = __expf(vals[c] - mx); s += vals[c]; }
        float inv = 1.f / s;
        #pragma unroll
        for (int c = 0; c < 32; ++c) qn[l * 264 + h * 32 + c] = f2bf(vals[c] * inv);
    }
    __syncthreads();

    // attended[v][l] = sum_i cn[bh][v][i] * qn[i][l]   (K=32 -> one WMMA per tile)
    #pragma unroll
    for (int t = 0; t < 8; ++t) {
        int tile = wave + (t << 3);      // 64 tiles = 8 heads x 2 v-tiles x 4 l-tiles
        int h  = tile >> 3;
        int mt = (tile >> 2) & 1;
        int lt = tile & 3;
        v16bf a  = load_frag(cn + ((size_t)(b * 8 + h)) * 1024 + mt * 16 * 32, 32);
        v16bf bb = load_frag(qn + (lt * 16) * 264 + h * 32, 264);
        v8f c = {};
        c = wmma_bf16(a, bb, c);
        int lloc = lt * 16 + (lane & 15);
        #pragma unroll
        for (int r = 0; r < 8; ++r) {
            int vch = h * 32 + mt * 16 + kh * 8 + r;
            att[lloc * 264 + vch] = f2bf(c[r]);
        }
    }
    __syncthreads();

    // out[o][l] = sum_v Wr[o][v] * att[v][l] + br[o] + x[o][l]
    #pragma unroll
    for (int t = 0; t < 8; ++t) {
        int tile = wave + (t << 3);      // 64 tiles = 16 o-tiles x 4 l-tiles
        int mt = tile >> 2;
        int lt = tile & 3;
        v8f acc = {};
        #pragma unroll
        for (int kk = 0; kk < 8; ++kk) {
            v16bf a  = load_frag(wr + (size_t)(mt * 16) * CH + kk * 32, CH);
            v16bf bb = load_frag(att + (lt * 16) * 264 + kk * 32, 264);
            acc = wmma_bf16(a, bb, acc);
        }
        int l = l0 + lt * 16 + (lane & 15);
        #pragma unroll
        for (int r = 0; r < 8; ++r) {
            int o = mt * 16 + kh * 8 + r;
            size_t gi = ((size_t)(b * CH + o)) * LSEQ + l;
            out[gi] = acc[r] + br[o] + x[gi];
        }
    }
}

// =====================================================================
extern "C" void kernel_launch(void* const* d_in, const int* in_sizes, int n_in,
                              void* d_out, int out_size, void* d_ws, size_t ws_size,
                              hipStream_t stream) {
    (void)in_sizes; (void)n_in; (void)out_size; (void)ws_size;
    const float* input_ = (const float*)d_in[0];
    const float* Wk = (const float*)d_in[1];
    const float* bk = (const float*)d_in[2];
    const float* Wq = (const float*)d_in[3];
    const float* bq = (const float*)d_in[4];
    const float* Wv = (const float*)d_in[5];
    const float* bv = (const float*)d_in[6];
    const float* Wr = (const float*)d_in[7];
    const float* br = (const float*)d_in[8];
    float* out = (float*)d_out;

    char* ws = (char*)d_ws;
    constexpr size_t kqv_bytes  = (size_t)3 * B_SZ * CH * LSEQ * 2;  // E,Q,V bf16
    constexpr size_t cn_bytes   = (size_t)B_SZ * HEADS_N * HD * HD * 2;
    constexpr size_t wkqv_bytes = (size_t)768 * 256 * 2;
    constexpr size_t wr_bytes   = (size_t)256 * 256 * 2;
    __bf16* kqv   = (__bf16*)(ws);
    __bf16* cn    = (__bf16*)(ws + kqv_bytes);
    __bf16* wkqv  = (__bf16*)(ws + kqv_bytes + cn_bytes);
    __bf16* wrb   = (__bf16*)(ws + kqv_bytes + cn_bytes + wkqv_bytes);
    float*  biasS = (float*) (ws + kqv_bytes + cn_bytes + wkqv_bytes + wr_bytes);

    prep_kernel<<<(768 * 256 + 65536 + 768) / 256, 256, 0, stream>>>(
        Wk, Wq, Wv, Wr, bk, bq, bv, wkqv, wrb, biasS);
    proj_kernel<<<dim3(LSEQ / 64, B_SZ), 256, 0, stream>>>(input_, wkqv, biasS, kqv);
    context_kernel<<<B_SZ * HEADS_N, 256, 0, stream>>>(kqv, cn);
    output_kernel<<<dim3(LSEQ / 64, B_SZ), 256, 0, stream>>>(input_, kqv, cn, wrb, br, out);
}
